// CausalSelfAttention_79508434583850
// MI455X (gfx1250) — matrix operand order repaired and checked by reading, not compiled, not run on hardware
//
#include <hip/hip_runtime.h>
#include <hip/hip_bf16.h>
#include <stdint.h>

typedef unsigned short u16;
typedef __attribute__((ext_vector_type(16))) __bf16 v16bf;
typedef __attribute__((ext_vector_type(8)))  __bf16 v8bf;
typedef __attribute__((ext_vector_type(8)))  float  v8f;

#define B_SZ 2
#define T_SZ 2048
#define C_SZ 1024
#define H_SZ 16
#define D_SZ 64
#define BT   (B_SZ * T_SZ)

// ---------- helpers ----------

__device__ __forceinline__ u16 f2bf(float f) {
  uint32_t u = __float_as_uint(f);
  u += 0x7FFFu + ((u >> 16) & 1u);     // round-to-nearest-even
  return (u16)(u >> 16);
}

__device__ __forceinline__ v16bf cat16(const u16* p0, const u16* p1) {
  v8bf lo = *reinterpret_cast<const v8bf*>(p0);   // 16B -> global/ds load_b128
  v8bf hi = *reinterpret_cast<const v8bf*>(p1);
  return __builtin_shufflevector(lo, hi, 0,1,2,3,4,5,6,7,8,9,10,11,12,13,14,15);
}

// A-matrix 16x32 bf16 tile from row-major memory (stride in elements).
// Lane l holds row (l&15); elems 0..7 -> K = half*8+0..7 ; elems 8..15 -> K = 16+half*8+0..7
__device__ __forceinline__ v16bf load_a(const u16* base, int stride, int lane) {
  const u16* p0 = base + (size_t)(lane & 15) * stride + (lane >> 4) * 8;
  return cat16(p0, p0 + 16);
}

// B-matrix 32x16 bf16 tile where element (k, n) lives at base[n*stride + k].
// Lane l holds column (l&15); elems 0..7 -> K = half*8+0..7 ; elems 8..15 -> K = 16+half*8+0..7
__device__ __forceinline__ v16bf load_b(const u16* base, int stride, int lane) {
  const u16* p = base + (size_t)(lane & 15) * stride + (lane >> 4) * 8;
  return cat16(p, p + 16);
}

__device__ __forceinline__ v8f wmma32(v16bf a, v16bf b, v8f c) {
  return __builtin_amdgcn_wmma_f32_16x16x32_bf16(false, a, false, b, (short)0, c, false, false);
}

// CDNA5 async copy global->LDS (ASYNCcnt-tracked), 16B per lane.
// VDST = VGPR holding wave-relative LDS byte address (flat addr low 32 bits).
__device__ __forceinline__ void async_b128(uint32_t lds_off, const u16* gptr) {
  asm volatile("global_load_async_to_lds_b128 %0, %1, off"
               :: "v"(lds_off), "v"(gptr) : "memory");
}

#define V8F_ZERO {0.0f,0.0f,0.0f,0.0f,0.0f,0.0f,0.0f,0.0f}

// ---------- kernel 0: f32 -> bf16 conversion ----------

__global__ void cvt_kernel(const float* __restrict__ src, u16* __restrict__ dst, int n) {
  int i = blockIdx.x * blockDim.x + threadIdx.x;
  if (i < n) dst[i] = f2bf(src[i]);
}

// ---------- kernel 1: fused Q/K/V projection ----------
// grid (BT/32, C/64), block 128 (4 waves). Wave computes a 32(M)x16(N) tile for q, k, v.
// q,k stored [B,H,T,D] bf16; v stored transposed [B,H,D,T] bf16.

__global__ __launch_bounds__(128) void qkv_proj_kernel(
    const u16* __restrict__ xbf,
    const u16* __restrict__ Wq, const u16* __restrict__ Wk, const u16* __restrict__ Wv,
    const float* __restrict__ bq, const float* __restrict__ bk, const float* __restrict__ bv,
    u16* __restrict__ qbuf, u16* __restrict__ kbuf, u16* __restrict__ vT)
{
  const int lane = threadIdx.x & 31;
  const int wave = threadIdx.x >> 5;
  const int m0 = blockIdx.x * 32;
  const int n0 = blockIdx.y * 64 + wave * 16;

  v8f aq0 = V8F_ZERO, aq1 = V8F_ZERO;
  v8f ak0 = V8F_ZERO, ak1 = V8F_ZERO;
  v8f av0 = V8F_ZERO, av1 = V8F_ZERO;

  for (int kk = 0; kk < C_SZ; kk += 32) {
    v16bf a0 = load_a(xbf + (size_t)m0 * C_SZ + kk, C_SZ, lane);
    v16bf a1 = load_a(xbf + (size_t)(m0 + 16) * C_SZ + kk, C_SZ, lane);
    v16bf wq = load_b(Wq + (size_t)n0 * C_SZ + kk, C_SZ, lane);  // B[k=c][n] = W[n][c]
    v16bf wk = load_b(Wk + (size_t)n0 * C_SZ + kk, C_SZ, lane);
    v16bf wv = load_b(Wv + (size_t)n0 * C_SZ + kk, C_SZ, lane);
    aq0 = wmma32(a0, wq, aq0);  aq1 = wmma32(a1, wq, aq1);
    ak0 = wmma32(a0, wk, ak0);  ak1 = wmma32(a1, wk, ak1);
    av0 = wmma32(a0, wv, av0);  av1 = wmma32(a1, wv, av1);
  }

  const int jcol = lane & 15, half = lane >> 4;
  const int n = n0 + jcol;
  const float bqv = bq[n], bkv = bk[n], bvv = bv[n];
  const int h = n >> 6, d = n & 63;

#pragma unroll
  for (int i = 0; i < 8; ++i) {
#pragma unroll
    for (int s = 0; s < 2; ++s) {
      int r = m0 + s * 16 + i + 8 * half;       // global row in [BT]
      int b = r >> 11;                          // / T_SZ
      int t = r & (T_SZ - 1);
      float qv = (s ? aq1[i] : aq0[i]) + bqv;
      float kv = (s ? ak1[i] : ak0[i]) + bkv;
      float vv = (s ? av1[i] : av0[i]) + bvv;
      size_t bh = (size_t)b * H_SZ + h;
      qbuf[(bh * T_SZ + t) * D_SZ + d] = f2bf(qv);
      kbuf[(bh * T_SZ + t) * D_SZ + d] = f2bf(kv);
      vT  [(bh * D_SZ + d) * T_SZ + t] = f2bf(vv);
    }
  }
}

// ---------- kernel 2: causal flash attention with async LDS staging ----------
// grid (B*H*(T/64)), block 128 (4 waves, 16 q rows each). K/V blocks of 32 keys are
// double-buffered in LDS via global_load_async_to_lds_b128; softmax + P@V per block.

__global__ __launch_bounds__(128) void attn_kernel(
    const u16* __restrict__ qbuf, const u16* __restrict__ kbuf,
    const u16* __restrict__ vT, u16* __restrict__ ybuf)
{
  __shared__ __align__(16) u16 ktile[2][32 * 64];   // [key 0..31][d 0..63]
  __shared__ __align__(16) u16 vtile[2][64 * 32];   // [d 0..63][key 0..31]
  __shared__ __align__(16) u16 plds_all[4][16 * 32];

  const int tid  = threadIdx.x;
  const int lane = tid & 31;
  const int wave = tid >> 5;
  u16* plds = plds_all[wave];

  const int qtiles = T_SZ / 64;
  const int qt = blockIdx.x % qtiles;
  const int bh = blockIdx.x / qtiles;

  const u16* qbase = qbuf + (size_t)bh * T_SZ * D_SZ;
  const u16* kbase = kbuf + (size_t)bh * T_SZ * D_SZ;
  const u16* vbase = vT   + (size_t)bh * D_SZ * T_SZ;

  const int i0 = qt * 64 + wave * 16;
  const int jcol = lane & 15, half = lane >> 4;

  // Q tile as two 16x32 A-operands (d = 0..31, 32..63), kept resident.
  v16bf qa0 = load_a(qbase + (size_t)i0 * D_SZ + 0,  D_SZ, lane);
  v16bf qa1 = load_a(qbase + (size_t)i0 * D_SZ + 32, D_SZ, lane);

  v8f o0 = V8F_ZERO, o1 = V8F_ZERO, o2 = V8F_ZERO, o3 = V8F_ZERO;
  float mrow[8], lrow[8];
#pragma unroll
  for (int i = 0; i < 8; ++i) { mrow[i] = -1.0e30f; lrow[i] = 0.0f; }

  const float sc = 0.125f;                  // 1/sqrt(D)
  const int jmaxw = i0 + 15;                // last key this wave needs (causal)
  const int nblocks = qt * 2 + 2;           // block-uniform trip count (keys 0..qt*64+63)

  // Cooperative stage of one 32-key K/V block: 4 async b128 per thread.
  auto stage_kv = [&](int jb, int buf) {
    const u16* kg = kbase + (size_t)jb * D_SZ;     // 32 rows x 64 d, contiguous 4KB
    u16* kl = ktile[buf];
    u16* vl = vtile[buf];
#pragma unroll
    for (int c = 0; c < 2; ++c) {
      int ck = tid * 2 + c;                        // 0..255 16B chunks
      async_b128((uint32_t)(uintptr_t)(kl + ck * 8), kg + (size_t)ck * 8);
      int row = ck >> 2, sub = ck & 3;             // V: 64 rows (d) x 4 chunks
      async_b128((uint32_t)(uintptr_t)(vl + row * 32 + sub * 8),
                 vbase + (size_t)row * T_SZ + jb + sub * 8);
    }
  };

  stage_kv(0, 0);
  int buf = 0;

  for (int ib = 0; ib < nblocks; ++ib) {
    const int jb = ib * 32;
    if (ib + 1 < nblocks) {
      stage_kv(jb + 32, buf ^ 1);                          // prefetch next block
      asm volatile("s_wait_asynccnt 0x4" ::: "memory");    // current block landed
    } else {
      asm volatile("s_wait_asynccnt 0x0" ::: "memory");
    }
    __syncthreads();

    if (jb <= jmaxw) {   // blocks fully above the causal frontier contribute nothing
      const u16* kl = ktile[buf];
      const u16* vl = vtile[buf];

      // ---- S = Q @ K^T for 32 keys (two 16x16 C tiles), B from LDS ----
      v8f s0 = V8F_ZERO, s1 = V8F_ZERO;
      s0 = wmma32(qa0, load_b(kl + 0,            D_SZ, lane), s0);
      s0 = wmma32(qa1, load_b(kl + 32,           D_SZ, lane), s0);
      s1 = wmma32(qa0, load_b(kl + 16 * 64,      D_SZ, lane), s1);
      s1 = wmma32(qa1, load_b(kl + 16 * 64 + 32, D_SZ, lane), s1);

      // ---- scale + causal mask + per-row block max ----
      float bmax[8];
#pragma unroll
      for (int i = 0; i < 8; ++i) {
        int qi = i0 + i + 8 * half;
        float x0 = s0[i] * sc; if (jb + jcol > qi)      x0 = -1.0e30f;
        float x1 = s1[i] * sc; if (jb + 16 + jcol > qi) x1 = -1.0e30f;
        s0[i] = x0; s1[i] = x1;
        bmax[i] = fmaxf(x0, x1);
      }
#pragma unroll
      for (int off = 1; off < 16; off <<= 1) {
#pragma unroll
        for (int i = 0; i < 8; ++i)
          bmax[i] = fmaxf(bmax[i], __shfl_xor(bmax[i], off, 32));
      }

      // ---- online softmax update; P -> LDS (C layout) ----
      float fac[8], rsum[8];
#pragma unroll
      for (int i = 0; i < 8; ++i) {
        float nm = fmaxf(mrow[i], bmax[i]);
        fac[i] = __expf(mrow[i] - nm);
        mrow[i] = nm;
        float p0 = __expf(s0[i] - nm);
        float p1 = __expf(s1[i] - nm);
        int m = i + 8 * half;
        plds[m * 32 + jcol]      = f2bf(p0);
        plds[m * 32 + 16 + jcol] = f2bf(p1);
        rsum[i] = p0 + p1;
      }
#pragma unroll
      for (int off = 1; off < 16; off <<= 1) {
#pragma unroll
        for (int i = 0; i < 8; ++i)
          rsum[i] += __shfl_xor(rsum[i], off, 32);
      }
#pragma unroll
      for (int i = 0; i < 8; ++i) {
        lrow[i] = lrow[i] * fac[i] + rsum[i];
        o0[i] *= fac[i]; o1[i] *= fac[i]; o2[i] *= fac[i]; o3[i] *= fac[i];
      }

      // ---- reload P in A-layout (CDNA5 split DS counter fence) ----
      asm volatile("s_wait_dscnt 0x0" ::: "memory");
      const u16* pp = plds + jcol * 32 + half * 8;
      v16bf pa = cat16(pp, pp + 16);

      // ---- O += P @ V, B from LDS (vtile rows contiguous along key) ----
      o0 = wmma32(pa, load_b(vl + 0 * 16 * 32, 32, lane), o0);
      o1 = wmma32(pa, load_b(vl + 1 * 16 * 32, 32, lane), o1);
      o2 = wmma32(pa, load_b(vl + 2 * 16 * 32, 32, lane), o2);
      o3 = wmma32(pa, load_b(vl + 3 * 16 * 32, 32, lane), o3);
    }

    __syncthreads();   // everyone done with buf before it is overwritten next iter
    buf ^= 1;
  }

  // ---- finalize: divide by row sum, store y bf16 [B*T, C] ----
  const int b = bh >> 4, h = bh & 15;
#pragma unroll
  for (int i = 0; i < 8; ++i) {
    float inv = 1.0f / lrow[i];
    int t = i0 + i + 8 * half;
    size_t r = ((size_t)b * T_SZ + t) * C_SZ + h * D_SZ + jcol;
    ybuf[r + 0]  = f2bf(o0[i] * inv);
    ybuf[r + 16] = f2bf(o1[i] * inv);
    ybuf[r + 32] = f2bf(o2[i] * inv);
    ybuf[r + 48] = f2bf(o3[i] * inv);
  }
}

// ---------- kernel 3: output projection (f32 out) ----------

__global__ __launch_bounds__(128) void out_proj_kernel(
    const u16* __restrict__ ybf, const u16* __restrict__ Wp,
    const float* __restrict__ bp, float* __restrict__ out)
{
  const int lane = threadIdx.x & 31;
  const int wave = threadIdx.x >> 5;
  const int m0 = blockIdx.x * 32;
  const int n0 = blockIdx.y * 64 + wave * 16;

  v8f a0 = V8F_ZERO, a1 = V8F_ZERO;
  for (int kk = 0; kk < C_SZ; kk += 32) {
    v16bf x0 = load_a(ybf + (size_t)m0 * C_SZ + kk, C_SZ, lane);
    v16bf x1 = load_a(ybf + (size_t)(m0 + 16) * C_SZ + kk, C_SZ, lane);
    v16bf w  = load_b(Wp + (size_t)n0 * C_SZ + kk, C_SZ, lane);
    a0 = wmma32(x0, w, a0);
    a1 = wmma32(x1, w, a1);
  }

  const int jcol = lane & 15, half = lane >> 4;
  const float bias = bp[n0 + jcol];
#pragma unroll
  for (int i = 0; i < 8; ++i) {
    int r0 = m0 + i + 8 * half;
    out[(size_t)r0 * C_SZ + n0 + jcol] = a0[i] + bias;
    out[(size_t)(r0 + 16) * C_SZ + n0 + jcol] = a1[i] + bias;
  }
}

// ---------- launch ----------

#define WS_MB(x) ((size_t)(x) << 20)

extern "C" void kernel_launch(void* const* d_in, const int* in_sizes, int n_in,
                              void* d_out, int out_size, void* d_ws, size_t ws_size,
                              hipStream_t stream) {
  const float* x  = (const float*)d_in[0];
  const float* Wq = (const float*)d_in[1];
  const float* bq = (const float*)d_in[2];
  const float* Wk = (const float*)d_in[3];
  const float* bk = (const float*)d_in[4];
  const float* Wv = (const float*)d_in[5];
  const float* bv = (const float*)d_in[6];
  const float* Wp = (const float*)d_in[7];
  const float* bp = (const float*)d_in[8];

  char* ws = (char*)d_ws;
  u16* xbf = (u16*)(ws + WS_MB(0));    // 8 MB  [BT, C] bf16
  u16* Wqb = (u16*)(ws + WS_MB(8));    // 2 MB
  u16* Wkb = (u16*)(ws + WS_MB(10));   // 2 MB
  u16* Wvb = (u16*)(ws + WS_MB(12));   // 2 MB
  u16* Wpb = (u16*)(ws + WS_MB(14));   // 2 MB
  u16* qb  = (u16*)(ws + WS_MB(16));   // 8 MB  [B,H,T,D]
  u16* kb  = (u16*)(ws + WS_MB(24));   // 8 MB  [B,H,T,D]
  u16* vtb = (u16*)(ws + WS_MB(32));   // 8 MB  [B,H,D,T]
  u16* yb  = (u16*)(ws + WS_MB(40));   // 8 MB  [BT, C]

  const int NX = BT * C_SZ;            // 4194304
  const int NW = C_SZ * C_SZ;          // 1048576
  cvt_kernel<<<(NX + 255) / 256, 256, 0, stream>>>(x,  xbf, NX);
  cvt_kernel<<<(NW + 255) / 256, 256, 0, stream>>>(Wq, Wqb, NW);
  cvt_kernel<<<(NW + 255) / 256, 256, 0, stream>>>(Wk, Wkb, NW);
  cvt_kernel<<<(NW + 255) / 256, 256, 0, stream>>>(Wv, Wvb, NW);
  cvt_kernel<<<(NW + 255) / 256, 256, 0, stream>>>(Wp, Wpb, NW);

  qkv_proj_kernel<<<dim3(BT / 32, C_SZ / 64), 128, 0, stream>>>(
      xbf, Wqb, Wkb, Wvb, bq, bk, bv, qb, kb, vtb);

  attn_kernel<<<dim3(B_SZ * H_SZ * (T_SZ / 64)), 128, 0, stream>>>(qb, kb, vtb, yb);

  out_proj_kernel<<<dim3(BT / 32, C_SZ / 64), 128, 0, stream>>>(
      yb, Wpb, bp, (float*)d_out);
}
